// ComplexGCN_14328010899641
// MI455X (gfx1250) — matrix operand order repaired and checked by reading, not compiled, not run on hardware
//
#include <hip/hip_runtime.h>
#include <hip/hip_bf16.h>

typedef __attribute__((ext_vector_type(16))) _Float16 v16h;
typedef __attribute__((ext_vector_type(8)))  _Float16 v8h;
typedef __attribute__((ext_vector_type(8)))  float    v8f;

#define GCN_N  4096
#define GCN_NE 65536

// ---------------------------------------------------------------------------
// WMMA fragment helpers (CDNA5 16x16x32 f16, layouts per ISA 7.12.2)
// ---------------------------------------------------------------------------

// A fragment: 16x32 tile of row-major A[lda], rows at row0, K at k0.
// lane: M = lane&15, half = lane>>4; elems 0..7 -> K = k0 + half*8 + e,
// elems 8..15 -> K = k0 + 16 + half*8 + (e-8). Both chunks contiguous (16B).
__device__ __forceinline__ v16h load_a_frag(const _Float16* A, int lda,
                                            int row0, int k0, int lane) {
  int m    = row0 + (lane & 15);
  int half = lane >> 4;
  const _Float16* p = A + (size_t)m * lda + k0 + half * 8;
  v8h lo = *(const v8h*)(p);
  v8h hi = *(const v8h*)(p + 16);
  return __builtin_shufflevector(lo, hi, 0,1,2,3,4,5,6,7,8,9,10,11,12,13,14,15);
}

// B fragment from TRANSPOSED operand Bt[N][K] (row-major, ldb):
// lane: N = n0 + (lane&15), half = lane>>4; elems 0..7 -> K = k0 + half*8 + e,
// elems 8..15 -> K = k0 + 16 + half*8 + (e-8).
__device__ __forceinline__ v16h load_b_frag(const _Float16* Bt, int ldb,
                                            int n0, int k0, int lane) {
  int n    = n0 + (lane & 15);
  int half = lane >> 4;
  const _Float16* p = Bt + (size_t)n * ldb + k0 + half * 8;
  v8h lo = *(const v8h*)(p);
  v8h hi = *(const v8h*)(p + 16);
  return __builtin_shufflevector(lo, hi, 0,1,2,3,4,5,6,7,8,9,10,11,12,13,14,15);
}

__device__ __forceinline__ v8f wmma16(v16h a, v16h b, v8f c) {
  return __builtin_amdgcn_wmma_f32_16x16x32_f16(false, a, false, b,
                                                (short)0, c, false, false);
}

// LDS byte offset of a shared-memory object: generic LDS addresses carry the
// aperture signature in bits [63:32] only; low 32 bits are the LDS offset.
__device__ __forceinline__ unsigned lds_off(const void* p) {
  return (unsigned)(uintptr_t)p;
}

// CDNA5 async copy: 16 bytes per lane, global -> LDS, tracked by ASYNCcnt.
__device__ __forceinline__ void async_ld16(unsigned ldsoff, const void* g) {
  asm volatile("global_load_async_to_lds_b128 %0, %1, off"
               :: "v"(ldsoff), "v"((unsigned long long)(uintptr_t)g)
               : "memory");
}

// ---------------------------------------------------------------------------
// GEMM: C[M][Nc] = A[M][K] * Bt[Nc][K]^T (+bias per col, optional relu)
// One wave per 16x64 output tile: A fragment reused across 4 B tiles.
// ---------------------------------------------------------------------------
__global__ __launch_bounds__(256)
void gemm_bt_kernel(const _Float16* __restrict__ A, const _Float16* __restrict__ Bt,
                    const float* __restrict__ bias, float* __restrict__ C,
                    int M, int Nc, int K, int relu) {
  int wave = (blockIdx.x * blockDim.x + threadIdx.x) >> 5;
  int lane = threadIdx.x & 31;
  int tilesN = Nc >> 6;                       // 64-wide output tiles
  int totalTiles = (M >> 4) * tilesN;
  if (wave >= totalTiles) return;             // wave-uniform: EXEC stays all-1s
  int tm  = (wave / tilesN) << 4;
  int tn0 = (wave % tilesN) << 6;
  v8f acc[4] = {v8f{}, v8f{}, v8f{}, v8f{}};
  for (int k = 0; k < K; k += 32) {
    v16h a  = load_a_frag(A, K, tm, k, lane);
    v16h b0 = load_b_frag(Bt, K, tn0,      k, lane);
    v16h b1 = load_b_frag(Bt, K, tn0 + 16, k, lane);
    v16h b2 = load_b_frag(Bt, K, tn0 + 32, k, lane);
    v16h b3 = load_b_frag(Bt, K, tn0 + 48, k, lane);
    acc[0] = wmma16(a, b0, acc[0]);
    acc[1] = wmma16(a, b1, acc[1]);
    acc[2] = wmma16(a, b2, acc[2]);
    acc[3] = wmma16(a, b3, acc[3]);
  }
  int cc    = lane & 15;
  int rbase = tm + ((lane >> 4) << 3);
#pragma unroll
  for (int t = 0; t < 4; ++t) {
    int col = tn0 + t * 16 + cc;
    float bv = bias ? bias[col] : 0.0f;
#pragma unroll
    for (int r = 0; r < 8; ++r) {
      float v = acc[t][r] + bv;
      if (relu) v = fmaxf(v, 0.0f);
      C[(size_t)(rbase + r) * Nc + col] = v;
    }
  }
}

// ---------------------------------------------------------------------------
// Flash attention, block-cooperative: a block (8 waves) owns one head and 8
// consecutive 16-query tiles. Each 32-key iteration stages the K tile (32x64)
// and Vt tile (64x32) into LDS with async global->LDS copies shared by all 8
// waves (8x less K/V global traffic), then computes with WMMA from LDS.
// Q [H][N][64] f16 (pre-scaled by 1/sqrt(D)), K [H][N][64] f16, Vt [H][64][N].
// Output: O16 [N][256] f16 (heads concatenated).
// ---------------------------------------------------------------------------
__global__ __launch_bounds__(256)
void flash_attn_kernel(const _Float16* __restrict__ Q, const _Float16* __restrict__ Kd,
                       const _Float16* __restrict__ Vt, _Float16* __restrict__ O16,
                       int Nn) {
  __shared__ _Float16 kbuf[32 * 64];          // K tile, row-major [key][d]
  __shared__ _Float16 vbuf[64 * 32];          // Vt tile, row-major [d][key]
  __shared__ _Float16 plds_all[8 * 16 * 32];  // per-wave prob staging
  int tid  = threadIdx.x;
  int wid  = tid >> 5;
  int lane = tid & 31;
  int blocksPerHead = Nn >> 7;                // 128 queries per block
  int h  = blockIdx.x / blocksPerHead;
  int q0 = (blockIdx.x % blocksPerHead) * 128 + wid * 16;
  const _Float16* Qh = Q  + (size_t)h * Nn * 64;
  const _Float16* Kh = Kd + (size_t)h * Nn * 64;
  const _Float16* Vh = Vt + (size_t)h * 64 * Nn;
  _Float16* plds = plds_all + wid * (16 * 32);

  v16h aq0 = load_a_frag(Qh, 64, q0, 0, lane);
  v16h aq1 = load_a_frag(Qh, 64, q0, 32, lane);

  v8f o0 = {}, o1 = {}, o2 = {}, o3 = {};
  float mrow[8], lrow[8];
#pragma unroll
  for (int r = 0; r < 8; ++r) { mrow[r] = -1e30f; lrow[r] = 0.0f; }

  int half = lane >> 4, c = lane & 15;
  // per-thread 16B staging assignments (256 threads cover each 4KB tile)
  int krow = tid >> 3, kcol = (tid & 7) * 8;  // kbuf[32][64]
  int vrow = tid >> 2, vcol = (tid & 3) * 8;  // vbuf[64][32]
  unsigned kdst = lds_off(&kbuf[krow * 64 + kcol]);
  unsigned vdst = lds_off(&vbuf[vrow * 32 + vcol]);

  for (int kt = 0; kt < Nn; kt += 32) {
    // async stage K/V tiles into LDS (ASYNCcnt), then block barrier
    async_ld16(kdst, (const void*)(Kh + (size_t)(kt + krow) * 64 + kcol));
    async_ld16(vdst, (const void*)(Vh + (size_t)vrow * Nn + kt + vcol));
    asm volatile("s_wait_asynccnt 0x0" ::: "memory");
    __syncthreads();

    // scores for 32 keys: two 16x16 tiles (K-dim = D = 64 -> 2 wmma steps each)
    v8f s0 = {}, s1 = {};
    s0 = wmma16(aq0, load_b_frag(kbuf, 64, 0,  0,  lane), s0);
    s0 = wmma16(aq1, load_b_frag(kbuf, 64, 0,  32, lane), s0);
    s1 = wmma16(aq0, load_b_frag(kbuf, 64, 16, 0,  lane), s1);
    s1 = wmma16(aq1, load_b_frag(kbuf, 64, 16, 32, lane), s1);

    float p0[8], p1[8];
#pragma unroll
    for (int r = 0; r < 8; ++r) {
      float v = fmaxf(s0[r], s1[r]);
#pragma unroll
      for (int m = 1; m < 16; m <<= 1) v = fmaxf(v, __shfl_xor(v, m, 32));
      float mn  = fmaxf(mrow[r], v);
      float fac = __expf(mrow[r] - mn);
      mrow[r] = mn;
      float e0 = __expf(s0[r] - mn);
      float e1 = __expf(s1[r] - mn);
      p0[r] = e0; p1[r] = e1;
      float sum = e0 + e1;
#pragma unroll
      for (int m = 1; m < 16; m <<= 1) sum += __shfl_xor(sum, m, 32);
      lrow[r] = lrow[r] * fac + sum;
      o0[r] *= fac; o1[r] *= fac; o2[r] *= fac; o3[r] *= fac;
    }

    // stage probabilities to LDS [16][32] f16, reload in A-fragment layout
#pragma unroll
    for (int r = 0; r < 8; ++r) {
      plds[(r + 8 * half) * 32 + c]      = (_Float16)p0[r];
      plds[(r + 8 * half) * 32 + 16 + c] = (_Float16)p1[r];
    }
    asm volatile("s_wait_dscnt 0" ::: "memory");
    v16h ap = load_a_frag(plds, 32, 0, 0, lane);

    // O(16x64) += P(16x32) @ V(32x64); vbuf rows are output cols -> contiguous
    o0 = wmma16(ap, load_b_frag(vbuf, 32, 0,  0, lane), o0);
    o1 = wmma16(ap, load_b_frag(vbuf, 32, 16, 0, lane), o1);
    o2 = wmma16(ap, load_b_frag(vbuf, 32, 32, 0, lane), o2);
    o3 = wmma16(ap, load_b_frag(vbuf, 32, 48, 0, lane), o3);

    __syncthreads();   // all waves done reading kbuf/vbuf before re-staging
  }

#pragma unroll
  for (int r = 0; r < 8; ++r) {
    float inv = 1.0f / lrow[r];
    int row = q0 + r + 8 * half;
    _Float16* dst = O16 + (size_t)row * 256 + h * 64 + c;
    dst[0]  = (_Float16)(o0[r] * inv);
    dst[16] = (_Float16)(o1[r] * inv);
    dst[32] = (_Float16)(o2[r] * inv);
    dst[48] = (_Float16)(o3[r] * inv);
  }
}

// ---------------------------------------------------------------------------
// Small utility kernels
// ---------------------------------------------------------------------------
__global__ void set_val_kernel(float* p, float v, int n) {
  int i = blockIdx.x * blockDim.x + threadIdx.x;
  if (i < n) p[i] = v;
}
__global__ void zero_kernel(float* p, int n) {
  int i = blockIdx.x * blockDim.x + threadIdx.x;
  if (i < n) p[i] = 0.0f;
}
__global__ void deg_accum_kernel(const int* __restrict__ col, const float* __restrict__ w,
                                 float* deg, int ne) {
  int e = blockIdx.x * blockDim.x + threadIdx.x;
  if (e < ne) atomicAdd(&deg[col[e]], w[e]);
}
__global__ void rsqrt_kernel(float* p, int n) {
  int i = blockIdx.x * blockDim.x + threadIdx.x;
  if (i < n) { float d = p[i]; p[i] = d > 0.0f ? rsqrtf(d) : 0.0f; }
}
__global__ void cvt16_kernel(const float* __restrict__ s, _Float16* __restrict__ d, int n) {
  int i = blockIdx.x * blockDim.x + threadIdx.x;
  if (i < n) d[i] = (_Float16)s[i];
}
// s[rows][cols] (f32) -> d[cols][rows] (f16)
__global__ void transpose_cvt_kernel(const float* __restrict__ s, _Float16* __restrict__ d,
                                     int rows, int cols) {
  int i = blockIdx.x * blockDim.x + threadIdx.x;
  if (i < rows * cols) {
    int r = i / cols, c = i % cols;
    d[(size_t)c * rows + r] = (_Float16)s[i];
  }
}
// Edge scatter: out[col] += m[row] * (dinv[row]*w*dinv[col]); one block per edge
__global__ void scatter_kernel(const float* __restrict__ m, const int* __restrict__ row,
                               const int* __restrict__ col, const float* __restrict__ w,
                               const float* __restrict__ dinv, float* __restrict__ out,
                               int dout) {
  int e = blockIdx.x;
  int r = row[e], cc = col[e];
  float norm = dinv[r] * w[e] * dinv[cc];
  for (int f = threadIdx.x; f < dout; f += blockDim.x)
    atomicAdd(&out[(size_t)cc * dout + f], m[(size_t)r * dout + f] * norm);
}
// h = relu(h + m*dinv^2 + bias); also emit f16 copy
__global__ void finalize_kernel(const float* __restrict__ m, const float* __restrict__ dinv,
                                const float* __restrict__ bias, float* __restrict__ h,
                                _Float16* __restrict__ h16, int n, int dout) {
  int i = blockIdx.x * blockDim.x + threadIdx.x;
  if (i >= n * dout) return;
  int node = i / dout, f = i % dout;
  float di = dinv[node];
  float v = h[i] + m[i] * di * di + bias[f];
  v = fmaxf(v, 0.0f);
  h[i] = v;
  h16[i] = (_Float16)v;
}
// qkv [N][768] f32 -> Q[H][N][64]*0.125, K[H][N][64], Vt[H][64][N] (all f16)
__global__ void split_qkv_kernel(const float* __restrict__ qkv, _Float16* __restrict__ q,
                                 _Float16* __restrict__ k, _Float16* __restrict__ v, int Nn) {
  int i = blockIdx.x * blockDim.x + threadIdx.x;
  if (i >= Nn * 256) return;
  int n = i >> 8, j = i & 255;
  int h = j >> 6, d = j & 63;
  const float* src = qkv + (size_t)n * 768;
  q[((size_t)h * Nn + n) * 64 + d] = (_Float16)(src[j] * 0.125f);
  k[((size_t)h * Nn + n) * 64 + d] = (_Float16)(src[256 + j]);
  v[((size_t)h * 64 + d) * Nn + n] = (_Float16)(src[512 + j]);
}
// LayerNorm per row; optional sigmoid; writes f32 and/or f16
__global__ void layernorm_kernel(const float* __restrict__ x, const float* __restrict__ g,
                                 const float* __restrict__ b, float* __restrict__ outf,
                                 _Float16* __restrict__ outh, int d, int do_sigmoid) {
  __shared__ float red[8];
  int rowi = blockIdx.x;
  const float* xr = x + (size_t)rowi * d;
  float v = (threadIdx.x < d) ? xr[threadIdx.x] : 0.0f;
  float s = v;
  for (int m = 1; m < 32; m <<= 1) s += __shfl_xor(s, m, 32);
  if ((threadIdx.x & 31) == 0) red[threadIdx.x >> 5] = s;
  __syncthreads();
  float tot = 0.0f;
  for (int i = 0; i < (int)(blockDim.x >> 5); ++i) tot += red[i];
  float mu = tot / d;
  __syncthreads();
  float dv = (threadIdx.x < d) ? (v - mu) : 0.0f;
  float s2 = dv * dv;
  for (int m = 1; m < 32; m <<= 1) s2 += __shfl_xor(s2, m, 32);
  if ((threadIdx.x & 31) == 0) red[threadIdx.x >> 5] = s2;
  __syncthreads();
  float tot2 = 0.0f;
  for (int i = 0; i < (int)(blockDim.x >> 5); ++i) tot2 += red[i];
  float var = tot2 / d;
  if (threadIdx.x < d) {
    float y = (v - mu) * rsqrtf(var + 1e-5f) * g[threadIdx.x] + b[threadIdx.x];
    if (do_sigmoid) y = 1.0f / (1.0f + __expf(-y));
    if (outf) outf[(size_t)rowi * d + threadIdx.x] = y;
    if (outh) outh[(size_t)rowi * d + threadIdx.x] = (_Float16)y;
  }
}

// ---------------------------------------------------------------------------
// Orchestration
// ---------------------------------------------------------------------------
extern "C" void kernel_launch(void* const* d_in, const int* in_sizes, int n_in,
                              void* d_out, int out_size, void* d_ws, size_t ws_size,
                              hipStream_t stream) {
  const int N = GCN_N, NE = GCN_NE;
  const float* xin = (const float*)d_in[0];
  const int*   ei  = (const int*)d_in[1];          // [2][NE]
  const float* ew  = (const float*)d_in[2];
  const float* W[4]    = {(const float*)d_in[3], (const float*)d_in[5],
                          (const float*)d_in[7], (const float*)d_in[9]};
  const float* Bb[4]   = {(const float*)d_in[4], (const float*)d_in[6],
                          (const float*)d_in[8], (const float*)d_in[10]};
  const float* WQKV[3] = {(const float*)d_in[11], (const float*)d_in[15], (const float*)d_in[19]};
  const float* BQKV[3] = {(const float*)d_in[12], (const float*)d_in[16], (const float*)d_in[20]};
  const float* WO[3]   = {(const float*)d_in[13], (const float*)d_in[17], (const float*)d_in[21]};
  const float* BO[3]   = {(const float*)d_in[14], (const float*)d_in[18], (const float*)d_in[22]};
  const float* G[4]    = {(const float*)d_in[23], (const float*)d_in[25],
                          (const float*)d_in[27], (const float*)d_in[29]};
  const float* BE[4]   = {(const float*)d_in[24], (const float*)d_in[26],
                          (const float*)d_in[28], (const float*)d_in[30]};

  char* wsp = (char*)d_ws;
  auto carve = [&](size_t bytes) -> char* {
    char* p = wsp; wsp += (bytes + 255) & ~(size_t)255; return p;
  };
  float*     dinv   = (float*)carve((size_t)N * 4);
  _Float16*  x16    = (_Float16*)carve((size_t)N * 256 * 2);
  _Float16*  w16    = (_Float16*)carve((size_t)768 * 256 * 2);
  float*     mbuf   = (float*)carve((size_t)N * 256 * 4);
  float*     hbuf   = (float*)carve((size_t)N * 256 * 4);
  _Float16*  h16    = (_Float16*)carve((size_t)N * 256 * 2);
  float*     qkv    = (float*)carve((size_t)N * 768 * 4);
  _Float16*  q16    = (_Float16*)carve((size_t)N * 256 * 2);
  _Float16*  k16    = (_Float16*)carve((size_t)N * 256 * 2);
  _Float16*  vt16   = (_Float16*)carve((size_t)N * 256 * 2);
  _Float16*  attn16 = (_Float16*)carve((size_t)N * 256 * 2);
  float*     proj   = (float*)carve((size_t)N * 256 * 4);

  // symmetric GCN normalization: deg = 1 (self-loop) + sum_{e: col==c} w[e]
  set_val_kernel<<<(N + 255) / 256, 256, 0, stream>>>(dinv, 1.0f, N);
  deg_accum_kernel<<<(NE + 255) / 256, 256, 0, stream>>>(ei + NE, ew, dinv, NE);
  rsqrt_kernel<<<(N + 255) / 256, 256, 0, stream>>>(dinv, N);
  cvt16_kernel<<<(N * 128 + 255) / 256, 256, 0, stream>>>(xin, x16, N * 128);

  const int din[4] = {128, 256, 256, 256};
  const int dou[4] = {256, 256, 256, 128};

  for (int l = 0; l < 4; ++l) {
    int K = din[l], Nc = dou[l];
    // m = x @ W  (Wt16 = W^T in f16 so GEMM fragments are contiguous)
    transpose_cvt_kernel<<<(K * Nc + 255) / 256, 256, 0, stream>>>(W[l], w16, K, Nc);
    int t1 = (N / 16) * (Nc / 64);
    gemm_bt_kernel<<<(t1 + 7) / 8, 256, 0, stream>>>(x16, w16, nullptr, mbuf, N, Nc, K, 0);
    // sparse aggregation + self-loop + bias + relu
    zero_kernel<<<(N * Nc + 255) / 256, 256, 0, stream>>>(hbuf, N * Nc);
    scatter_kernel<<<NE, 256, 0, stream>>>(mbuf, ei, ei + NE, ew, dinv, hbuf, Nc);
    finalize_kernel<<<(N * Nc + 255) / 256, 256, 0, stream>>>(mbuf, dinv, Bb[l], hbuf, h16, N, Nc);

    if (l < 3) {
      // qkv = x @ wqkv^T + bqkv   (wqkv stored [768,256] == Bt layout already)
      cvt16_kernel<<<(768 * 256 + 255) / 256, 256, 0, stream>>>(WQKV[l], w16, 768 * 256);
      int t2 = (N / 16) * (768 / 64);
      gemm_bt_kernel<<<(t2 + 7) / 8, 256, 0, stream>>>(h16, w16, BQKV[l], qkv, N, 768, 256, 0);
      split_qkv_kernel<<<(N * 256 + 255) / 256, 256, 0, stream>>>(qkv, q16, k16, vt16, N);
      flash_attn_kernel<<<4 * (N / 128), 256, 0, stream>>>(q16, k16, vt16, attn16, N);
      // out proj: o @ wo^T + bo   (wo stored [256,256] == Bt layout)
      cvt16_kernel<<<(256 * 256 + 255) / 256, 256, 0, stream>>>(WO[l], w16, 256 * 256);
      int t3 = (N / 16) * (256 / 64);
      gemm_bt_kernel<<<(t3 + 7) / 8, 256, 0, stream>>>(attn16, w16, BO[l], proj, N, 256, 256, 0);
      // LayerNorm -> next layer's f16 input
      layernorm_kernel<<<N, 256, 0, stream>>>(proj, G[l], BE[l], nullptr, x16, 256, 0);
    } else {
      // final LN + sigmoid -> d_out (f32, [N,128])
      layernorm_kernel<<<N, 128, 0, stream>>>(hbuf, G[3], BE[3], (float*)d_out, nullptr, 128, 1);
    }
  }
}
